// DilatedReparamBlock_84018150244706
// MI455X (gfx1250) — matrix operand order repaired and checked by reading, not compiled, not run on hardware
//
#include <hip/hip_runtime.h>
#include <hip/hip_bf16.h>
#include <math.h>

typedef __bf16  bf16_t;
typedef __bf16  v8bf  __attribute__((ext_vector_type(8)));
typedef __bf16  v16bf __attribute__((ext_vector_type(16)));
typedef float   v8f   __attribute__((ext_vector_type(8)));

#define C_DIM 384
#define HW    3136
#define WIDTH 56
#define EPSV  1e-5f

// Pack two floats as adjacent bf16 halves of a u32 (element k in low 16 bits).
__device__ __forceinline__ unsigned pack2_hi(float a, float b) {
  unsigned short ua = __builtin_bit_cast(unsigned short, (bf16_t)a);
  unsigned short ub = __builtin_bit_cast(unsigned short, (bf16_t)b);
  return (unsigned)ua | ((unsigned)ub << 16);
}
__device__ __forceinline__ unsigned pack2_lo(float a, float b) {
  float ra = a - (float)(bf16_t)a;
  float rb = b - (float)(bf16_t)b;
  unsigned short ua = __builtin_bit_cast(unsigned short, (bf16_t)ra);
  unsigned short ub = __builtin_bit_cast(unsigned short, (bf16_t)rb);
  return (unsigned)ua | ((unsigned)ub << 16);
}

// ---------------- Kernel 1: pointwise GEMM (3xBF16 split) + BN + residual ---
#define BM 128
#define BN 64
#define BK 32

__global__ __launch_bounds__(256) void pw_gemm_bn_res(
    const float* __restrict__ x,  const float* __restrict__ w,
    const float* __restrict__ g,  const float* __restrict__ bb,
    const float* __restrict__ mm, const float* __restrict__ vv,
    float* __restrict__ xpw)
{
  __shared__ alignas(16) bf16_t lAh[BM * BK];
  __shared__ alignas(16) bf16_t lAl[BM * BK];
  __shared__ alignas(16) bf16_t lBh[BN * BK];
  __shared__ alignas(16) bf16_t lBl[BN * BK];
  __shared__ float ldsSc[BM];
  __shared__ float ldsSh[BM];

  unsigned* const uAh = (unsigned*)lAh;
  unsigned* const uAl = (unsigned*)lAl;
  unsigned* const uBh = (unsigned*)lBh;
  unsigned* const uBl = (unsigned*)lBl;

  const int t    = threadIdx.x;
  const int lane = t & 31;
  const int wv   = t >> 5;
  const int n0   = blockIdx.x * BN;   // hw tile
  const int m0   = blockIdx.y * BM;   // output-channel tile
  const int b    = blockIdx.z;

  const int wm = (wv & 3) * 32;       // wave M offset in block tile
  const int wn = (wv >> 2) * 32;      // wave N offset in block tile

  // Per-channel BN scale/shift for this block's 128 output channels (once).
  if (t < BM) {
    const int o = m0 + t;
    const float sc = g[o] * rsqrtf(vv[o] + EPSV);
    ldsSc[t] = sc;
    ldsSh[t] = bb[o] - mm[o] * sc;
  }

  v8f acc[2][2];
  {
    v8f z = {};
    for (int i = 0; i < 2; i++)
      for (int j = 0; j < 2; j++) acc[i][j] = z;
  }

  const int kA0  = (lane >= 16) ? 8 : 0;   // A-fragment K base (16-bit A 16x32)
  const int kB0  = (lane >= 16) ? 8 : 0;   // B-fragment K base (32x16 B)
  const int rsel = lane & 15;

  for (int k0 = 0; k0 < C_DIM; k0 += BK) {
    // Stage W tile [BM x BK]: one float2 global load -> one b32 DS store/plane.
    #pragma unroll
    for (int j = 0; j < (BM * BK / 2) / 256; ++j) {     // 8 iters
      int idx = j * 256 + t;                            // pair index
      int kp  = idx & 15;                               // k-pair (k = 2*kp)
      int mi  = idx >> 4;
      const float2 f2 = *(const float2*)&w[(m0 + mi) * C_DIM + k0 + 2 * kp];
      uAh[idx] = pack2_hi(f2.x, f2.y);                  // uint idx == mi*16+kp
      uAl[idx] = pack2_lo(f2.x, f2.y);
    }
    // Stage X tile [BK x BN] transposed to [n][k]: two strided (coalesced-in-n)
    // loads -> one b32 DS store per plane.
    #pragma unroll
    for (int j = 0; j < (BK * BN / 2) / 256; ++j) {     // 4 iters
      int idx = j * 256 + t;
      int ni  = idx & 63;
      int kp  = idx >> 6;                               // k = 2*kp
      const float* p = &x[((size_t)b * C_DIM + k0 + 2 * kp) * HW + n0 + ni];
      if (k0 + BK < C_DIM) __builtin_prefetch(p + (size_t)BK * HW, 0, 1);
      const float f0 = p[0];
      const float f1 = p[HW];
      uBh[ni * 16 + kp] = pack2_hi(f0, f1);
      uBl[ni * 16 + kp] = pack2_lo(f0, f1);
    }
    __syncthreads();

    #pragma unroll
    for (int mi = 0; mi < 2; mi++) {
      const int arow = (wm + mi * 16 + rsel) * BK;
      union { v16bf v; v8bf h[2]; } ah, al;
      ah.h[0] = *(const v8bf*)&lAh[arow + kA0];
      ah.h[1] = *(const v8bf*)&lAh[arow + kA0 + 16];
      al.h[0] = *(const v8bf*)&lAl[arow + kA0];
      al.h[1] = *(const v8bf*)&lAl[arow + kA0 + 16];
      #pragma unroll
      for (int ni = 0; ni < 2; ni++) {
        const int brow = (wn + ni * 16 + rsel) * BK;
        union { v16bf v; v8bf h[2]; } bh, bl;
        bh.h[0] = *(const v8bf*)&lBh[brow + kB0];
        bh.h[1] = *(const v8bf*)&lBh[brow + kB0 + 16];
        bl.h[0] = *(const v8bf*)&lBl[brow + kB0];
        bl.h[1] = *(const v8bf*)&lBl[brow + kB0 + 16];
        v8f c = acc[mi][ni];
        c = __builtin_amdgcn_wmma_f32_16x16x32_bf16(false, ah.v, false, bh.v,
                                                    (short)0, c, false, false);
        c = __builtin_amdgcn_wmma_f32_16x16x32_bf16(false, ah.v, false, bl.v,
                                                    (short)0, c, false, false);
        c = __builtin_amdgcn_wmma_f32_16x16x32_bf16(false, al.v, false, bh.v,
                                                    (short)0, c, false, false);
        acc[mi][ni] = c;
      }
    }
    __syncthreads();
  }

  // Epilogue: BN + residual. C/D layout: VGPR r -> M = r (+8 upper half-wave),
  // N = lane & 15. Scale/shift come from LDS (one pair per channel).
  const int rhalf = (lane >= 16) ? 8 : 0;
  #pragma unroll
  for (int mi = 0; mi < 2; mi++) {
    #pragma unroll
    for (int ni = 0; ni < 2; ni++) {
      const int col = n0 + wn + ni * 16 + rsel;
      #pragma unroll
      for (int r = 0; r < 8; r++) {
        const int ol = wm + mi * 16 + r + rhalf;        // channel within block
        const size_t off = ((size_t)b * C_DIM + m0 + ol) * HW + col;
        xpw[off] = acc[mi][ni][r] * ldsSc[ol] + ldsSh[ol] + x[off];
      }
    }
  }
}

// ---------------- Kernel 2: depthwise 13x13 + BN + 1x1-dw branch + exact GELU
#define TS   28
#define HALO 6
#define LW   (TS + 2 * HALO)   // 40

__global__ __launch_bounds__(256) void dwconv_bn_gelu(
    const float* __restrict__ xpw,
    const float* __restrict__ kw,
    const float* __restrict__ kg,  const float* __restrict__ kb,
    const float* __restrict__ km,  const float* __restrict__ kv,
    const float* __restrict__ w1,
    const float* __restrict__ g1,  const float* __restrict__ b1,
    const float* __restrict__ m1,  const float* __restrict__ v1,
    float* __restrict__ out)
{
  __shared__ float tile[LW * LW];
  __shared__ float wk[169];

  const int t  = threadIdx.x;
  const int tx = (blockIdx.x & 1) * TS;
  const int ty = (blockIdx.x >> 1) * TS;
  const int c  = blockIdx.y;
  const int b  = blockIdx.z;

  const size_t base = ((size_t)b * C_DIM + c) * HW;

  // halo tile load (zero pad outside image)
  for (int idx = t; idx < LW * LW; idx += 256) {
    const int ly = idx / LW;
    const int lx = idx - ly * LW;
    const int gy = ty + ly - HALO;
    const int gx = tx + lx - HALO;
    float v = 0.0f;
    if (gy >= 0 && gy < WIDTH && gx >= 0 && gx < WIDTH)
      v = xpw[base + (size_t)gy * WIDTH + gx];
    tile[idx] = v;
  }
  if (t < 169) wk[t] = kw[c * 169 + t];

  // block-uniform BN / branch constants (channel uniform -> scalar loads)
  const float scK = kg[c] * rsqrtf(kv[c] + EPSV);
  const float shK = kb[c] - km[c] * scK;
  const float sc1 = g1[c] * rsqrtf(v1[c] + EPSV);
  const float s1w = w1[c] * sc1;
  const float sh1 = b1[c] - m1[c] * sc1;

  __syncthreads();

  for (int oidx = t; oidx < TS * TS; oidx += 256) {
    const int oy = oidx / TS;
    const int ox = oidx - oy * TS;
    float s = 0.0f;
    #pragma unroll
    for (int ky = 0; ky < 13; ky++) {
      const float* row = &tile[(oy + ky) * LW + ox];
      const float* wr  = &wk[ky * 13];
      #pragma unroll
      for (int kx = 0; kx < 13; kx++) s = fmaf(row[kx], wr[kx], s);
    }
    const float xc = tile[(oy + HALO) * LW + ox + HALO];
    const float v  = s * scK + shK + xc * s1w + sh1;
    const float ge = 0.5f * v * (1.0f + erff(v * 0.70710678118654752f));
    out[base + (size_t)(ty + oy) * WIDTH + (tx + ox)] = ge;
  }
}

// ---------------------------------------------------------------------------
extern "C" void kernel_launch(void* const* d_in, const int* in_sizes, int n_in,
                              void* d_out, int out_size, void* d_ws, size_t ws_size,
                              hipStream_t stream) {
  const float* x     = (const float*)d_in[0];
  const float* pw_w  = (const float*)d_in[1];
  const float* pw_g  = (const float*)d_in[2];
  const float* pw_b  = (const float*)d_in[3];
  const float* pw_m  = (const float*)d_in[4];
  const float* pw_v  = (const float*)d_in[5];
  const float* dwk_w = (const float*)d_in[6];
  const float* dwk_g = (const float*)d_in[7];
  const float* dwk_b = (const float*)d_in[8];
  const float* dwk_m = (const float*)d_in[9];
  const float* dwk_v = (const float*)d_in[10];
  const float* dw1_w = (const float*)d_in[11];
  const float* dw1_g = (const float*)d_in[12];
  const float* dw1_b = (const float*)d_in[13];
  const float* dw1_m = (const float*)d_in[14];
  const float* dw1_v = (const float*)d_in[15];

  float* xpw = (float*)d_ws;          // 16*384*3136 f32 = ~77 MB scratch
  float* out = (float*)d_out;

  dim3 g1(HW / BN, C_DIM / BM, 16);   // (49, 3, 16)
  pw_gemm_bn_res<<<g1, 256, 0, stream>>>(x, pw_w, pw_g, pw_b, pw_m, pw_v, xpw);

  dim3 g2((WIDTH / TS) * (WIDTH / TS), C_DIM, 16);  // (4, 384, 16)
  dwconv_bn_gelu<<<g2, 256, 0, stream>>>(xpw, dwk_w, dwk_g, dwk_b, dwk_m, dwk_v,
                                         dw1_w, dw1_g, dw1_b, dw1_m, dw1_v, out);
}
